// CurricularFace_22806276342384
// MI455X (gfx1250) — compile-verified, matrix operand order checked by reading, not run on hardware
//
#include <hip/hip_runtime.h>
#include <math.h>

// ---------------- problem constants ----------------
#define NROW 512
#define FEAT 512
#define NCLS 100000
#define NCB  782            // ceil(NCLS / 128) column blocks in the GEMM
#define S_SCALE 64.0f
#define COS_M_C 0.8775825618903728f
#define SIN_M_C 0.479425538604203f
#define THRESH_C (-0.8775825618903728f)
#define MM_C 0.23971276930210156f
#define SHIFT_C 66.0f       // fixed log-sum-exp shift: |logit| <= ~64.7

typedef __attribute__((ext_vector_type(16))) __bf16 v16bf;
typedef __attribute__((ext_vector_type(8)))  float  v8f;

union FragBF { uint4 u[2]; v16bf v; };

// bf16 helpers via bit ops (round-to-nearest-even)
static __device__ __forceinline__ unsigned short f2bf(float f) {
    unsigned int u = __float_as_uint(f);
    unsigned int r = u + 0x7fffu + ((u >> 16) & 1u);
    return (unsigned short)(r >> 16);
}
static __device__ __forceinline__ float bf2f(unsigned short h) {
    return __uint_as_float(((unsigned int)h) << 16);
}

// ---------------- kernel 1: row-normalize E, split to bf16 hi/lo ----------------
__global__ void enorm_kernel(const float* __restrict__ emb,
                             unsigned short* __restrict__ ehi,
                             unsigned short* __restrict__ elo) {
    __shared__ float red[256];
    int row = blockIdx.x;
    int tid = threadIdx.x;
    float v0 = emb[row * FEAT + tid];
    float v1 = emb[row * FEAT + tid + 256];
    red[tid] = v0 * v0 + v1 * v1;
    __syncthreads();
    for (int s = 128; s > 0; s >>= 1) {
        if (tid < s) red[tid] += red[tid + s];
        __syncthreads();
    }
    float rn = rsqrtf(red[0]);
    float f0 = v0 * rn, f1 = v1 * rn;
    unsigned short h0 = f2bf(f0), h1 = f2bf(f1);
    ehi[row * FEAT + tid]       = h0;
    ehi[row * FEAT + tid + 256] = h1;
    elo[row * FEAT + tid]       = f2bf(f0 - bf2f(h0));
    elo[row * FEAT + tid + 256] = f2bf(f1 - bf2f(h1));
}

// ---------------- kernel 2: target logits (512 small dot products) ----------------
// Computes its own per-label column norm: tl = clamp(dot(e, w_col) * rsqrt(sum w_col^2))
__global__ void target_kernel(const unsigned short* __restrict__ ehi,
                              const unsigned short* __restrict__ elo,
                              const float* __restrict__ w,
                              const int* __restrict__ label,
                              float* __restrict__ tl) {
    int lane = threadIdx.x & 31;
    int wid  = threadIdx.x >> 5;
    int row  = blockIdx.x * 8 + wid;
    int lab  = label[row];
    float acc = 0.0f, ssq = 0.0f;
    for (int k = lane; k < FEAT; k += 32) {
        float e  = bf2f(ehi[row * FEAT + k]) + bf2f(elo[row * FEAT + k]);
        float wv = w[(size_t)k * NCLS + lab];
        acc = fmaf(e, wv, acc);
        ssq = fmaf(wv, wv, ssq);
    }
    for (int m = 16; m >= 1; m >>= 1) {
        acc += __shfl_xor(acc, m);
        ssq += __shfl_xor(ssq, m);
    }
    if (lane == 0) {
        float c = acc * rsqrtf(ssq);
        tl[row] = fminf(fmaxf(c, -1.0f), 1.0f);
    }
}

// ---------------- kernel 3: margin params + scalar t ----------------
__global__ void stats_kernel(const float* __restrict__ tl,
                             float* __restrict__ cosm,
                             float* __restrict__ flt,
                             float* __restrict__ tparam) {
    __shared__ float red[512];
    int tid = threadIdx.x;
    float v = tl[tid];
    red[tid] = v;
    __syncthreads();
    for (int s = 256; s > 0; s >>= 1) {
        if (tid < s) red[tid] += red[tid + s];
        __syncthreads();
    }
    if (tid == 0) tparam[0] = 0.01f * (red[0] / (float)NROW);
    float sn = sqrtf(fmaxf(0.0f, 1.0f - v * v));
    float cm = v * COS_M_C - sn * SIN_M_C;
    cosm[tid] = cm;
    flt[tid]  = (v > THRESH_C) ? cm : (v - MM_C);
}

// ---------------- kernel 4: fused bf16x3 WMMA GEMM + softmax partials ----------------
// Grid: NCB blocks of 1024 threads (32 waves). Block tile: 512 rows x 128 cols.
// Wave grid 8(M) x 4(N): wave tile 64 rows x 32 cols = 4x2 WMMA tiles.
// - W (raw fp32) is read from HBM exactly once across the whole dispatch.
// - Column sum-of-squares accumulated during staging; rsqrt applied in epilogue.
// - Double-buffered LDS B tile; one barrier per K-step; next W tile prefetched
//   into registers while the current step's 24 WMMAs execute.
__global__ __launch_bounds__(1024)
void gemm_softmax_kernel(const float* __restrict__ w,
                         const unsigned short* __restrict__ ehi,
                         const unsigned short* __restrict__ elo,
                         const float* __restrict__ cosm,
                         const float* __restrict__ flt,
                         const int* __restrict__ label,
                         const float* __restrict__ tparam,
                         float* __restrict__ pexp,
                         float* __restrict__ plog) {
    __shared__ __align__(16) unsigned short sBhi[2][128 * 40];
    __shared__ __align__(16) unsigned short sBlo[2][128 * 40];
    __shared__ float ssqred[128 * 8];
    __shared__ float rcol[128];
    __shared__ float redE[NROW][4];
    __shared__ float redL[NROW][4];

    const int tid  = threadIdx.x;
    const int lane = tid & 31;
    const int wid  = tid >> 5;     // 0..31
    const int wm   = wid & 7;      // row group: rows wm*64 .. +63
    const int wn   = wid >> 3;     // col group: cols wn*32 .. +31
    const int col0 = blockIdx.x * 128;

    const int lmod = lane & 15;
    const int lgrp = lane >> 4;    // 0/1
    const int koff = lgrp ? 8 : 0; // K sub-offset per ISA 16-bit A/B layout

    // B staging mapping: thread -> (col cB in 0..127, k row kB in 0..7), 4 passes
    const int cB = tid & 127;
    const int kB = tid >> 7;
    const bool colvalid = (col0 + cB) < NCLS;

    v8f acc[4][2];
    const v8f vzero = {0.f, 0.f, 0.f, 0.f, 0.f, 0.f, 0.f, 0.f};
#pragma unroll
    for (int mi = 0; mi < 4; ++mi)
#pragma unroll
        for (int ni = 0; ni < 2; ++ni) acc[mi][ni] = vzero;

    // preload W tile for k0 = 0 into registers
    float wv[4];
#pragma unroll
    for (int p = 0; p < 4; ++p)
        wv[p] = colvalid ? w[(size_t)(kB + p * 8) * NCLS + (col0 + cB)] : 0.0f;

    float ssq = 0.0f;  // this thread's partial sum of squares for column cB

    for (int it = 0; it < FEAT / 32; ++it) {
        const int k0  = it * 32;
        const int buf = it & 1;

        // convert + store current raw-W tile (bf16 hi/lo, transposed [col][k]);
        // fold in the column sum-of-squares on the fly
#pragma unroll
        for (int p = 0; p < 4; ++p) {
            int k = kB + p * 8;
            float s = wv[p];
            ssq = fmaf(s, s, ssq);
            unsigned short h = f2bf(s);
            sBhi[buf][cB * 40 + k] = h;
            sBlo[buf][cB * 40 + k] = f2bf(s - bf2f(h));
        }
        __syncthreads();

        // prefetch next W tile into registers (hidden behind this step's WMMAs)
        if (it + 1 < FEAT / 32) {
#pragma unroll
            for (int p = 0; p < 4; ++p)
                wv[p] = colvalid
                          ? w[(size_t)(k0 + 32 + kB + p * 8) * NCLS + (col0 + cB)]
                          : 0.0f;
        }

        FragBF bhi[2], blo[2];
#pragma unroll
        for (int ni = 0; ni < 2; ++ni) {
            int cc = wn * 32 + ni * 16 + lmod;
            const unsigned short* ph = &sBhi[buf][cc * 40 + koff];
            bhi[ni].u[0] = *(const uint4*)(ph);
            bhi[ni].u[1] = *(const uint4*)(ph + 16);
            const unsigned short* pl = &sBlo[buf][cc * 40 + koff];
            blo[ni].u[0] = *(const uint4*)(pl);
            blo[ni].u[1] = *(const uint4*)(pl + 16);
        }
#pragma unroll
        for (int mi = 0; mi < 4; ++mi) {
            int rA = wm * 64 + mi * 16 + lmod;
            FragBF ah, al;
            const unsigned short* pa = &ehi[rA * FEAT + k0 + koff];
            ah.u[0] = *(const uint4*)(pa);
            ah.u[1] = *(const uint4*)(pa + 16);
            const unsigned short* pb = &elo[rA * FEAT + k0 + koff];
            al.u[0] = *(const uint4*)(pb);
            al.u[1] = *(const uint4*)(pb + 16);
#pragma unroll
            for (int ni = 0; ni < 2; ++ni) {
                acc[mi][ni] = __builtin_amdgcn_wmma_f32_16x16x32_bf16(
                    false, ah.v, false, bhi[ni].v, (short)0, acc[mi][ni], false, false);
                acc[mi][ni] = __builtin_amdgcn_wmma_f32_16x16x32_bf16(
                    false, ah.v, false, blo[ni].v, (short)0, acc[mi][ni], false, false);
                acc[mi][ni] = __builtin_amdgcn_wmma_f32_16x16x32_bf16(
                    false, al.v, false, bhi[ni].v, (short)0, acc[mi][ni], false, false);
            }
        }
    }

    // ---------------- per-column rsqrt(sum w^2) ----------------
    ssqred[cB * 8 + kB] = ssq;
    __syncthreads();
    if (tid < 128) {
        float s = 0.0f;
#pragma unroll
        for (int j = 0; j < 8; ++j) s += ssqred[tid * 8 + j];
        rcol[tid] = ((col0 + tid) < NCLS) ? rsqrtf(s) : 0.0f;  // 0 avoids 0*inf NaN
    }
    __syncthreads();

    // ---------------- epilogue: transform + per-row partial sums ----------------
    const float t = tparam[0];
#pragma unroll
    for (int mi = 0; mi < 4; ++mi) {
        int rbase = wm * 64 + mi * 16 + 8 * lgrp;
        float esum[8], lsum[8];
#pragma unroll
        for (int v = 0; v < 8; ++v) { esum[v] = 0.0f; lsum[v] = 0.0f; }
#pragma unroll
        for (int ni = 0; ni < 2; ++ni) {
            int colg = col0 + wn * 32 + ni * 16 + lmod;
            bool ok = colg < NCLS;
            float rc = rcol[wn * 32 + ni * 16 + lmod];
#pragma unroll
            for (int v = 0; v < 8; ++v) {
                int rowg = rbase + v;
                float c = acc[mi][ni][v] * rc;
                c = fminf(fmaxf(c, -1.0f), 1.0f);
                float cp;
                if (colg == label[rowg]) {
                    cp = flt[rowg];
                } else {
                    float cm = cosm[rowg];
                    cp = (c > cm) ? c * (t + c) : c;
                }
                float logit = S_SCALE * cp;
                if (ok) {
                    esum[v] += __expf(logit - SHIFT_C);
                    lsum[v] += logit;
                }
            }
        }
#pragma unroll
        for (int v = 0; v < 8; ++v) {
            float e = esum[v], lg = lsum[v];
            // reduce across the 16 lanes holding the same row (stay inside half-wave)
            for (int m = 8; m >= 1; m >>= 1) {
                e  += __shfl_xor(e, m);
                lg += __shfl_xor(lg, m);
            }
            if (lmod == 0) {
                redE[rbase + v][wn] = e;
                redL[rbase + v][wn] = lg;
            }
        }
    }
    __syncthreads();
    if (tid < NROW) {
        float E = redE[tid][0] + redE[tid][1] + redE[tid][2] + redE[tid][3];
        float L = redL[tid][0] + redL[tid][1] + redL[tid][2] + redL[tid][3];
        pexp[(size_t)tid * NCB + blockIdx.x] = E;
        plog[(size_t)tid * NCB + blockIdx.x] = L;
    }
}

// ---------------- kernel 5: per-row reduction -> per-row loss ----------------
__global__ void rowreduce_kernel(const float* __restrict__ pexp,
                                 const float* __restrict__ plog,
                                 const float* __restrict__ flt,
                                 float* __restrict__ lossrow) {
    __shared__ float rE[256];
    __shared__ float rL[256];
    int row = blockIdx.x;
    int tid = threadIdx.x;
    float E = 0.0f, L = 0.0f;
    for (int j = tid; j < NCB; j += 256) {
        E += pexp[(size_t)row * NCB + j];
        L += plog[(size_t)row * NCB + j];
    }
    rE[tid] = E; rL[tid] = L;
    __syncthreads();
    for (int s = 128; s > 0; s >>= 1) {
        if (tid < s) { rE[tid] += rE[tid + s]; rL[tid] += rL[tid + s]; }
        __syncthreads();
    }
    if (tid == 0) {
        float lse    = SHIFT_C + logf(rE[0]);
        float nll    = lse - S_SCALE * flt[row];
        float smooth = lse - rL[0] / (float)NCLS;
        lossrow[row] = 0.9f * nll + 0.1f * smooth;
    }
}

// ---------------- kernel 6: final mean ----------------
__global__ void finloss_kernel(const float* __restrict__ lossrow, float* __restrict__ out) {
    __shared__ float red[512];
    int tid = threadIdx.x;
    red[tid] = lossrow[tid];
    __syncthreads();
    for (int s = 256; s > 0; s >>= 1) {
        if (tid < s) red[tid] += red[tid + s];
        __syncthreads();
    }
    if (tid == 0) out[0] = red[0] / (float)NROW;
}

// ---------------- workspace layout (bytes) ----------------
#define OFF_EHI     0ull                       // ushort[512*512]  524288 B
#define OFF_ELO     524288ull                  // ushort[512*512]  524288 B
#define OFF_TL      1048576ull                 // float[512]
#define OFF_COSM    1050624ull                 // float[512]
#define OFF_FLT     1052672ull                 // float[512]
#define OFF_T       1054720ull                 // float[1]
#define OFF_PEXP    1054976ull                 // float[512*NCB]   1601536 B
#define OFF_PLOG    2656512ull                 // float[512*NCB]   1601536 B
#define OFF_LOSSROW 4258048ull                 // float[512]

extern "C" void kernel_launch(void* const* d_in, const int* in_sizes, int n_in,
                              void* d_out, int out_size, void* d_ws, size_t ws_size,
                              hipStream_t stream) {
    const float* emb   = (const float*)d_in[0];   // [512, 512]
    const float* w     = (const float*)d_in[1];   // [512, 100000]
    const int*   label = (const int*)d_in[2];     // [512]
    float* out = (float*)d_out;

    char* ws = (char*)d_ws;
    unsigned short* ehi     = (unsigned short*)(ws + OFF_EHI);
    unsigned short* elo     = (unsigned short*)(ws + OFF_ELO);
    float*          tl      = (float*)(ws + OFF_TL);
    float*          cosm    = (float*)(ws + OFF_COSM);
    float*          flt     = (float*)(ws + OFF_FLT);
    float*          tparam  = (float*)(ws + OFF_T);
    float*          pexp    = (float*)(ws + OFF_PEXP);
    float*          plog    = (float*)(ws + OFF_PLOG);
    float*          lossrow = (float*)(ws + OFF_LOSSROW);

    enorm_kernel<<<dim3(NROW), dim3(256), 0, stream>>>(emb, ehi, elo);
    target_kernel<<<dim3(NROW / 8), dim3(256), 0, stream>>>(ehi, elo, w, label, tl);
    stats_kernel<<<dim3(1), dim3(512), 0, stream>>>(tl, cosm, flt, tparam);
    gemm_softmax_kernel<<<dim3(NCB), dim3(1024), 0, stream>>>(
        w, ehi, elo, cosm, flt, label, tparam, pexp, plog);
    rowreduce_kernel<<<dim3(NROW), dim3(256), 0, stream>>>(pexp, plog, flt, lossrow);
    finloss_kernel<<<dim3(1), dim3(512), 0, stream>>>(lossrow, out);
}
